// CAD_Coarse_Grained_13211319403312
// MI455X (gfx1250) — matrix operand order repaired and checked by reading, not compiled, not run on hardware
//
#include <hip/hip_runtime.h>
#include <hip/hip_bf16.h>

// ---- CDNA5 WMMA vector types -------------------------------------------------
typedef __bf16 bf16_t;
typedef bf16_t v16bf __attribute__((ext_vector_type(16)));
typedef float  v8f   __attribute__((ext_vector_type(8)));
typedef unsigned int v4u __attribute__((ext_vector_type(4)));
typedef unsigned int v2u __attribute__((ext_vector_type(2)));

#define DDIM    1536
#define BM      128
#define BN      128
#define BK      32
#define KSTEPS  (DDIM / BK)    // 48
#define PTOT    4096
#define NCHUNKS (PTOT / BN)    // 32
#define MTOT    25088          // 8 * 3136
#define NBLK    (MTOT / BM)    // 196
#define ST      40             // padded LDS row stride (bf16 elems); 80B -> conflict-free

union FragBF { v16bf v; v4u q[2]; };

// fp32 -> bf16 split: hi = truncate(f) (exact remainder), lo = RNE(f - hi)
__device__ __forceinline__ unsigned short bf_trunc(float f) {
    return (unsigned short)(__float_as_uint(f) >> 16);
}
__device__ __forceinline__ unsigned short bf_rne(float f) {
    unsigned u = __float_as_uint(f);
    u += 0x7FFFu + ((u >> 16) & 1u);
    return (unsigned short)(u >> 16);
}
__device__ __forceinline__ float bf2f(unsigned short h) {
    return __uint_as_float(((unsigned)h) << 16);
}
__device__ __forceinline__ void split2(float f, unsigned short& h, unsigned short& l) {
    h = bf_trunc(f);
    l = bf_rne(f - bf2f(h));   // f - hi is exact in fp32
}
__device__ __forceinline__ void stage4(float4 v, unsigned short* hiP, unsigned short* loP) {
    unsigned short h0,l0,h1,l1,h2,l2,h3,l3;
    split2(v.x,h0,l0); split2(v.y,h1,l1); split2(v.z,h2,l2); split2(v.w,h3,l3);
    v2u hh = { (unsigned)h0 | ((unsigned)h1 << 16), (unsigned)h2 | ((unsigned)h3 << 16) };
    v2u ll = { (unsigned)l0 | ((unsigned)l1 << 16), (unsigned)l2 | ((unsigned)l3 << 16) };
    *(v2u*)hiP = hh;
    *(v2u*)loP = ll;
}

// ---- Kernel 1: centroid squared norms ---------------------------------------
__global__ void cnorm_kernel(const float* __restrict__ centroids,
                             float* __restrict__ cnorm) {
    const int wave = threadIdx.x >> 5;
    const int lane = threadIdx.x & 31;
    const int c = blockIdx.x * 8 + wave;                 // 512 blocks * 8 waves = 4096
    const float4* p = (const float4*)(centroids + (size_t)c * DDIM);
    float s = 0.f;
    #pragma unroll
    for (int j = 0; j < 12; ++j) {                       // 12 * 32 lanes * 4 = 1536
        float4 v = p[lane + j * 32];
        s += v.x*v.x + v.y*v.y + v.z*v.z + v.w*v.w;
    }
    #pragma unroll
    for (int m = 16; m >= 1; m >>= 1) s += __shfl_xor(s, m, 32);
    if (lane == 0) cnorm[c] = s;
}

// ---- Kernel 2: min_c ||x - c|| via bf16x3 WMMA GEMM + row-min ----------------
__global__ __launch_bounds__(256) void knn_min_dist_kernel(
        const float* __restrict__ embeds,
        const float* __restrict__ centroids,
        const float* __restrict__ cnorm,
        float* __restrict__ out) {

    __shared__ unsigned short sAhi[BM * ST];
    __shared__ unsigned short sAlo[BM * ST];
    __shared__ unsigned short sBhi[BN * ST];
    __shared__ unsigned short sBlo[BN * ST];
    __shared__ float xnormS[BM];
    __shared__ float cnormS[BN];
    __shared__ int   rowMin[BM];

    const int tid  = threadIdx.x;
    const int lane = tid & 31;
    const int wave = tid >> 5;       // 0..7
    const int l15  = lane & 15;
    const int lh   = lane >> 4;      // lane half: 0 or 1
    const int wm   = wave >> 2;      // 0..1  -> M offset *64
    const int wn   = wave & 3;       // 0..3  -> N offset *32

    const size_t rowBase = (size_t)blockIdx.x * BM;

    if (tid < BM) rowMin[tid] = 0x7F800000;              // +inf bits

    // per-row ||x||^2 : each wave handles 16 rows, coalesced strided float4
    for (int i = 0; i < 16; ++i) {
        size_t row = rowBase + (size_t)wave * 16 + i;
        const float4* p = (const float4*)(embeds + row * DDIM);
        float s = 0.f;
        #pragma unroll
        for (int j = 0; j < 12; ++j) {
            float4 v = p[lane + j * 32];
            s += v.x*v.x + v.y*v.y + v.z*v.z + v.w*v.w;
        }
        #pragma unroll
        for (int m = 16; m >= 1; m >>= 1) s += __shfl_xor(s, m, 32);
        if (lane == 0) xnormS[wave * 16 + i] = s;
    }
    __syncthreads();

    const v8f vzero = {0.f,0.f,0.f,0.f,0.f,0.f,0.f,0.f};

    for (int nc = 0; nc < NCHUNKS; ++nc) {
        __syncthreads();                                  // protect cnormS + tiles reuse
        if (tid < BN) cnormS[tid] = cnorm[nc * BN + tid];

        v8f acc[4][2];
        #pragma unroll
        for (int mf = 0; mf < 4; ++mf)
            #pragma unroll
            for (int nf = 0; nf < 2; ++nf)
                acc[mf][nf] = vzero;

        for (int ks = 0; ks < KSTEPS; ++ks) {
            __syncthreads();                              // prev frag reads done
            // ---- stage A(128x32) and B(128x32) fp32 -> hi/lo bf16 in LDS ----
            const float4* Ag = (const float4*)(embeds + rowBase * DDIM);
            const float4* Bg = (const float4*)(centroids + (size_t)nc * BN * DDIM);
            #pragma unroll
            for (int i = 0; i < 4; ++i) {
                int linear = tid + i * 256;               // 0..1023
                int r  = linear >> 3;                     // row 0..127
                int k4 = linear & 7;                      // float4 within K-chunk
                float4 av = Ag[(size_t)r * (DDIM/4) + ks * 8 + k4];
                float4 bv = Bg[(size_t)r * (DDIM/4) + ks * 8 + k4];
                int base = r * ST + k4 * 4;
                stage4(av, &sAhi[base], &sAlo[base]);
                stage4(bv, &sBhi[base], &sBlo[base]);
            }
            __syncthreads();

            // ---- fragment loads (ISA 16-bit A 16x32 layout; B per-lane column) ----
            FragBF ahi[4], alo[4], bhi[2], blo[2];
            const int kA0 = lh ? 8  : 0;                  // lanes 0-15: K0-7,  16-31: K8-15
            const int kA1 = lh ? 24 : 16;                 // lanes 0-15: K16-23,16-31: K24-31
            #pragma unroll
            for (int mf = 0; mf < 4; ++mf) {
                int m = wm * 64 + mf * 16 + l15;
                ahi[mf].q[0] = *(const v4u*)&sAhi[m * ST + kA0];
                ahi[mf].q[1] = *(const v4u*)&sAhi[m * ST + kA1];
                alo[mf].q[0] = *(const v4u*)&sAlo[m * ST + kA0];
                alo[mf].q[1] = *(const v4u*)&sAlo[m * ST + kA1];
            }
            const int kB0 = lh ? 8 : 0;                   // lanes 0-15: K0-7,K16-23; 16-31: K8-15,K24-31
            #pragma unroll
            for (int nf = 0; nf < 2; ++nf) {
                int n = wn * 32 + nf * 16 + l15;
                bhi[nf].q[0] = *(const v4u*)&sBhi[n * ST + kB0];
                bhi[nf].q[1] = *(const v4u*)&sBhi[n * ST + kB0 + 16];
                blo[nf].q[0] = *(const v4u*)&sBlo[n * ST + kB0];
                blo[nf].q[1] = *(const v4u*)&sBlo[n * ST + kB0 + 16];
            }

            // ---- bf16x3: dot ~= hi*hi + hi*lo + lo*hi (fp32 accumulate) ----
            #pragma unroll
            for (int mf = 0; mf < 4; ++mf)
                #pragma unroll
                for (int nf = 0; nf < 2; ++nf) {
                    acc[mf][nf] = __builtin_amdgcn_wmma_f32_16x16x32_bf16(
                        false, ahi[mf].v, false, bhi[nf].v, (short)0, acc[mf][nf], false, false);
                    acc[mf][nf] = __builtin_amdgcn_wmma_f32_16x16x32_bf16(
                        false, ahi[mf].v, false, blo[nf].v, (short)0, acc[mf][nf], false, false);
                    acc[mf][nf] = __builtin_amdgcn_wmma_f32_16x16x32_bf16(
                        false, alo[mf].v, false, bhi[nf].v, (short)0, acc[mf][nf], false, false);
                }
        }

        // ---- epilogue: dist^2 = ||x||^2 + ||c||^2 - 2*dot, running row-min ----
        // C layout: VGPR r -> lanes 0-15: M=r, N=lane; lanes 16-31: M=r+8, N=lane-16
        #pragma unroll
        for (int nf = 0; nf < 2; ++nf) {
            float cn = cnormS[wn * 32 + nf * 16 + l15];
            #pragma unroll
            for (int mf = 0; mf < 4; ++mf) {
                #pragma unroll
                for (int r = 0; r < 8; ++r) {
                    int mrow = wm * 64 + mf * 16 + lh * 8 + r;
                    float v = xnormS[mrow] + cn - 2.0f * acc[mf][nf][r];
                    v = fmaxf(v, 0.0f);
                    // min across the 16 lanes sharing this M (xor stays in half)
                    v = fminf(v, __shfl_xor(v, 1, 32));
                    v = fminf(v, __shfl_xor(v, 2, 32));
                    v = fminf(v, __shfl_xor(v, 4, 32));
                    v = fminf(v, __shfl_xor(v, 8, 32));
                    if (l15 == 0)
                        atomicMin(&rowMin[mrow], __float_as_int(v)); // nonneg floats: int order ok
                }
            }
        }
    }

    __syncthreads();
    if (tid < BM) {
        float d2 = fmaxf(__int_as_float(rowMin[tid]), 0.0f);
        out[1 + rowBase + tid] = sqrtf(d2);               // score, flat [B*N]
    }
    if (blockIdx.x == 0 && tid == 0) out[0] = 0.0f;       // loss (eval mode)
}

// ---- launcher ----------------------------------------------------------------
extern "C" void kernel_launch(void* const* d_in, const int* in_sizes, int n_in,
                              void* d_out, int out_size, void* d_ws, size_t ws_size,
                              hipStream_t stream) {
    const float* embeds    = (const float*)d_in[0];   // [8, 3136, 1536] fp32
    const float* centroids = (const float*)d_in[1];   // [4096, 1536]    fp32
    float* out   = (float*)d_out;                     // [1 + 25088] fp32
    float* cnorm = (float*)d_ws;                      // 4096 floats scratch (16 KB)

    hipLaunchKernelGGL(cnorm_kernel, dim3(PTOT / 8), dim3(256), 0, stream,
                       centroids, cnorm);
    hipLaunchKernelGGL(knn_min_dist_kernel, dim3(NBLK), dim3(256), 0, stream,
                       embeds, centroids, cnorm, out);
}
